// MultiKR_21268678050244
// MI455X (gfx1250) — matrix operand order repaired and checked by reading, not compiled, not run on hardware
//
#include <hip/hip_runtime.h>

// MultiKR fused forward for MI455X (gfx1250, wave32).
// Roofline: ~4.3 GFLOP vs ~25 MB gather traffic (~170 FLOP/B) -> GEMMs run on
// v_wmma_f32_16x16x32_f16 (f16 in, f32 accumulate). One fused kernel keeps all
// activations in LDS f16 per 128-row tile; weights pre-converted to f16 in d_ws.
// This rev: lrelu as v_max_num_f32 (max(v, 0.01v)), vectorized f16 LDS stores.

#define B_TOTAL 16384
#define D 128
#define H1 256
#define H2 128
#define TM 128          // batch rows per workgroup
#define STRIDE 264      // padded LDS row stride (halfs): 528 B, 16B-aligned, bank-spread

typedef __attribute__((ext_vector_type(16))) _Float16 v16h;
typedef __attribute__((ext_vector_type(8)))  _Float16 v8h;
typedef __attribute__((ext_vector_type(4)))  _Float16 v4h;
typedef __attribute__((ext_vector_type(8)))  float    v8f;

// leaky_relu(v) == max(v, 0.01*v) elementwise -> single v_max_num_f32
__device__ __forceinline__ float lrelu(float v) { return fmaxf(v, 0.01f * v); }

__device__ __forceinline__ float wave_sum(float v) {
#pragma unroll
  for (int off = 16; off > 0; off >>= 1) v += __shfl_xor(v, off, 32);
  return v;
}

// A fragment (16x32 f16, MxK) from LDS, row-major with 'stride' halfs.
// ISA 7.12.2 layout: lane&15 = row; lanes>=16 hold the upper 8-K interleave;
// VGPRs 0-3 cover K base+[0..15], VGPRs 4-7 cover K base+16+[0..15].
__device__ __forceinline__ v16h load_a_lds(const _Float16* base, int stride,
                                           int m0, int kb, int lane) {
  int r  = m0 + (lane & 15);
  int hs = (lane >> 4) * 8;
  const _Float16* p = base + r * stride + kb + hs;
  v8h lo = *reinterpret_cast<const v8h*>(p);
  v8h hi = *reinterpret_cast<const v8h*>(p + 16);
  v16h a;
#pragma unroll
  for (int i = 0; i < 8; ++i) { a[i] = lo[i]; a[i + 8] = hi[i]; }
  return a;
}

// B fragment (32x16 f16, KxN) from global f16 weight stored [N][K] row-major
// (torch-style W for out = x @ W^T). Lane&15 = column N; lanes>=16 hold the upper 8-K interleave.
__device__ __forceinline__ v16h load_b_w(const _Float16* __restrict__ W, int K,
                                         int n0, int kb, int lane) {
  int n = n0 + (lane & 15);
  const _Float16* p = W + n * K + kb + (lane >> 4) * 8;
  v8h lo = *reinterpret_cast<const v8h*>(p);
  v8h hi = *reinterpret_cast<const v8h*>(p + 16);
  v16h b;
#pragma unroll
  for (int i = 0; i < 8; ++i) { b[i] = lo[i]; b[i + 8] = hi[i]; }
  return b;
}

// C/D 16x16 f32 layout: VGPR r -> M = r + 8*(lane>>4), N = lane&15.
// Fused bias + leaky-relu, write back to LDS as f16 for the next stage.
__device__ __forceinline__ void store_tile(_Float16* dst, int stride, int m0, int n0,
                                           const v8f& acc,
                                           const float* __restrict__ bias, int lane) {
  int n  = n0 + (lane & 15);
  float bv = bias[n];
  int mo = m0 + ((lane >> 4) << 3);
#pragma unroll
  for (int r = 0; r < 8; ++r) {
    float v = lrelu(acc[r] + bv);
    dst[(mo + r) * stride + n] = (_Float16)v;
  }
}

__global__ void convert_weights(const float* __restrict__ ulow,
                                const float* __restrict__ W0,
                                const float* __restrict__ W1,
                                _Float16* __restrict__ ws) {
  const int N0 = D * D;          // 16384
  const int N1 = H1 * 2 * D;     // 65536
  const int N2 = H2 * H1;        // 32768
  int i = blockIdx.x * blockDim.x + threadIdx.x;
  if (i < N0)                 ws[i] = (_Float16)ulow[i];
  else if (i < N0 + N1)       ws[i] = (_Float16)W0[i - N0];
  else if (i < N0 + N1 + N2)  ws[i] = (_Float16)W1[i - N0 - N1];
}

__global__ __launch_bounds__(256)
void multikr_fused(const int*   __restrict__ user_ids,
                   const int*   __restrict__ item_ids,
                   const float* __restrict__ rec_target,
                   const float* __restrict__ user_table,
                   const float* __restrict__ item_table,
                   const float* __restrict__ entity_table,
                   const float* __restrict__ w_vv, const float* __restrict__ w_ev,
                   const float* __restrict__ w_ve, const float* __restrict__ w_ee,
                   const float* __restrict__ b_v,  const float* __restrict__ b_e,
                   const float* __restrict__ ulow_b,
                   const float* __restrict__ rec_b0,
                   const float* __restrict__ rec_b1,
                   const float* __restrict__ rec_W2,
                   const float* __restrict__ rec_b2,
                   const _Float16* __restrict__ wsUlow,  // [D][D]    f16
                   const _Float16* __restrict__ wsW0,    // [H1][2D]  f16
                   const _Float16* __restrict__ wsW1,    // [H2][H1]  f16
                   float* __restrict__ out) {
  extern __shared__ __align__(16) _Float16 lds[];
  _Float16* xbuf  = lds;                 // [TM][STRIDE]: 0..127 user act, 128..255 item; later h2 in 0..127
  _Float16* h1buf = lds + TM * STRIDE;   // [TM][STRIDE]: 0..255 h1

  const int tid   = threadIdx.x;
  const int lane  = tid & 31;
  const int w     = tid >> 5;            // wave id 0..7; wave owns rows 16w..16w+15
  const int rBase = blockIdx.x * TM;

  // ---- Phase 1: cooperative gather of user rows -> f16 xbuf[:, 0:128] ----
#pragma unroll
  for (int i = 0; i < 16; ++i) {
    int idx = tid + i * 256;             // 0..4095 = 128 rows * 32 float4
    int r   = idx >> 5;
    int c4  = idx & 31;
    int g   = user_ids[rBase + r];
    const float4 v = reinterpret_cast<const float4*>(user_table)[g * 32 + c4];
    v4h h; h[0] = (_Float16)v.x; h[1] = (_Float16)v.y;
           h[2] = (_Float16)v.z; h[3] = (_Float16)v.w;
    *reinterpret_cast<v4h*>(xbuf + r * STRIDE + c4 * 4) = h;  // ds_store_b64
  }

  // ---- Phase 2: cross-compress (wave-private rows; no xbuf[0:128] dependence) ----
  {
    const float4 wvv = reinterpret_cast<const float4*>(w_vv)[lane];
    const float4 wev = reinterpret_cast<const float4*>(w_ev)[lane];
    const float4 wve = reinterpret_cast<const float4*>(w_ve)[lane];
    const float4 wee = reinterpret_cast<const float4*>(w_ee)[lane];
    const float bv = b_v[0], be = b_e[0];
#pragma unroll 1
    for (int i = 0; i < 16; ++i) {
      int r = w * 16 + i;
      int g = item_ids[rBase + r];
      float4 it = reinterpret_cast<const float4*>(item_table)[g * 32 + lane];
      float4 hd = reinterpret_cast<const float4*>(entity_table)[g * 32 + lane];
#pragma unroll
      for (int l = 0; l < 2; ++l) {
        float svv = wave_sum(hd.x * wvv.x + hd.y * wvv.y + hd.z * wvv.z + hd.w * wvv.w);
        float sev = wave_sum(it.x * wev.x + it.y * wev.y + it.z * wev.z + it.w * wev.w);
        float sve = wave_sum(hd.x * wve.x + hd.y * wve.y + hd.z * wve.z + hd.w * wve.w);
        float see = wave_sum(it.x * wee.x + it.y * wee.y + it.z * wee.z + it.w * wee.w);
        float4 ni, nh;
        ni.x = it.x * svv + hd.x * sev + bv;  nh.x = it.x * sve + hd.x * see + be;
        ni.y = it.y * svv + hd.y * sev + bv;  nh.y = it.y * sve + hd.y * see + be;
        ni.z = it.z * svv + hd.z * sev + bv;  nh.z = it.z * sve + hd.z * see + be;
        ni.w = it.w * svv + hd.w * sev + bv;  nh.w = it.w * sve + hd.w * see + be;
        it = ni; hd = nh;
      }
      v4h h; h[0] = (_Float16)it.x; h[1] = (_Float16)it.y;
             h[2] = (_Float16)it.z; h[3] = (_Float16)it.w;
      *reinterpret_cast<v4h*>(xbuf + r * STRIDE + 128 + lane * 4) = h;
    }
  }

  __syncthreads();  // only barrier: phase-1 gather was block-wide; everything after is wave-private

  // ---- Phase 3: user MLP, 2x: user = lrelu(user @ ulow_W^T + b) ----
#pragma unroll 1
  for (int layer = 0; layer < 2; ++layer) {
    v16h aw[4];
#pragma unroll
    for (int kk = 0; kk < 4; ++kk) aw[kk] = load_a_lds(xbuf, STRIDE, w * 16, kk * 32, lane);
#pragma unroll 1
    for (int tn = 0; tn < 8; ++tn) {
      v8f acc = {};
#pragma unroll
      for (int kk = 0; kk < 4; ++kk) {
        v16h bw = load_b_w(wsUlow, D, tn * 16, kk * 32, lane);
        acc = __builtin_amdgcn_wmma_f32_16x16x32_f16(false, aw[kk], false, bw,
                                                     (short)0, acc, false, false);
      }
      store_tile(xbuf, STRIDE, w * 16, tn * 16, acc, ulow_b, lane);
    }
  }

  // ---- Phase 4: rec0: h1 = lrelu([user,item] @ W0^T + b0), K=256, N=256 ----
  {
    v16h aw[8];
#pragma unroll
    for (int kk = 0; kk < 8; ++kk) aw[kk] = load_a_lds(xbuf, STRIDE, w * 16, kk * 32, lane);
#pragma unroll 1
    for (int tn = 0; tn < 16; ++tn) {
      v8f acc = {};
#pragma unroll
      for (int kk = 0; kk < 8; ++kk) {
        v16h bw = load_b_w(wsW0, 2 * D, tn * 16, kk * 32, lane);
        acc = __builtin_amdgcn_wmma_f32_16x16x32_f16(false, aw[kk], false, bw,
                                                     (short)0, acc, false, false);
      }
      store_tile(h1buf, STRIDE, w * 16, tn * 16, acc, rec_b0, lane);
    }
  }

  // ---- Phase 5: rec1: h2 = lrelu(h1 @ W1^T + b1), K=256, N=128 -> xbuf[:,0:128] ----
  {
    v16h aw[8];
#pragma unroll
    for (int kk = 0; kk < 8; ++kk) aw[kk] = load_a_lds(h1buf, STRIDE, w * 16, kk * 32, lane);
#pragma unroll 1
    for (int tn = 0; tn < 8; ++tn) {
      v8f acc = {};
#pragma unroll
      for (int kk = 0; kk < 8; ++kk) {
        v16h bw = load_b_w(wsW1, H1, tn * 16, kk * 32, lane);
        acc = __builtin_amdgcn_wmma_f32_16x16x32_f16(false, aw[kk], false, bw,
                                                     (short)0, acc, false, false);
      }
      store_tile(xbuf, STRIDE, w * 16, tn * 16, acc, rec_b1, lane);
    }
  }

  // ---- Phase 6: out = lrelu(h2 . w2 + b2); plus rec_target passthrough ----
  {
    const float4 w2 = reinterpret_cast<const float4*>(rec_W2)[lane];
    const float  b2 = rec_b2[0];
#pragma unroll 1
    for (int i = 0; i < 16; ++i) {
      int r = w * 16 + i;
      const _Float16* p = xbuf + r * STRIDE + lane * 4;
      v4h h = *reinterpret_cast<const v4h*>(p);
      float s = (float)h[0] * w2.x + (float)h[1] * w2.y +
                (float)h[2] * w2.z + (float)h[3] * w2.w;
      s = wave_sum(s);
      if (lane == 0) out[rBase + r] = lrelu(s + b2);
    }
  }
  if (tid < TM) out[B_TOTAL + rBase + tid] = rec_target[rBase + tid];
}

extern "C" void kernel_launch(void* const* d_in, const int* in_sizes, int n_in,
                              void* d_out, int out_size, void* d_ws, size_t ws_size,
                              hipStream_t stream) {
  const int*   user_ids     = (const int*)d_in[0];
  const int*   item_ids     = (const int*)d_in[1];
  const float* rec_target   = (const float*)d_in[2];
  const float* user_table   = (const float*)d_in[3];
  const float* item_table   = (const float*)d_in[4];
  const float* entity_table = (const float*)d_in[5];
  const float* w_vv   = (const float*)d_in[6];
  const float* w_ev   = (const float*)d_in[7];
  const float* w_ve   = (const float*)d_in[8];
  const float* w_ee   = (const float*)d_in[9];
  const float* b_v    = (const float*)d_in[10];
  const float* b_e    = (const float*)d_in[11];
  const float* ulow_W = (const float*)d_in[12];
  const float* ulow_b = (const float*)d_in[13];
  const float* rec_W0 = (const float*)d_in[14];
  const float* rec_b0 = (const float*)d_in[15];
  const float* rec_W1 = (const float*)d_in[16];
  const float* rec_b1 = (const float*)d_in[17];
  const float* rec_W2 = (const float*)d_in[18];
  const float* rec_b2 = (const float*)d_in[19];
  float* out = (float*)d_out;

  _Float16* ws = (_Float16*)d_ws;           // f16 weight copies: ulow | W0 | W1
  _Float16* wsUlow = ws;
  _Float16* wsW0   = ws + D * D;
  _Float16* wsW1   = ws + D * D + H1 * 2 * D;

  const int NCVT = D * D + H1 * 2 * D + H2 * H1;   // 114688
  convert_weights<<<(NCVT + 255) / 256, 256, 0, stream>>>(ulow_W, rec_W0, rec_W1, ws);

  const size_t ldsBytes = (size_t)2 * TM * STRIDE * sizeof(_Float16);  // 135168 B (<320 KB/WGP)
  hipFuncSetAttribute((const void*)multikr_fused,
                      hipFuncAttributeMaxDynamicSharedMemorySize, (int)ldsBytes);
  multikr_fused<<<B_TOTAL / TM, 256, ldsBytes, stream>>>(
      user_ids, item_ids, rec_target, user_table, item_table, entity_table,
      w_vv, w_ev, w_ve, w_ee, b_v, b_e,
      ulow_b, rec_b0, rec_b1, rec_W2, rec_b2,
      wsUlow, wsW0, wsW1, out);
}
